// GroupedMultiQueryAttention_78245714199294
// MI455X (gfx1250) — matrix operand order repaired and checked by reading, not compiled, not run on hardware
//
#include <hip/hip_runtime.h>
#include <hip/hip_bf16.h>

typedef __attribute__((ext_vector_type(16))) _Float16 v16h;
typedef __attribute__((ext_vector_type(8)))  float    v8f;

union Frag { uint4 u[2]; v16h h; };
union H8   { uint4 u;    _Float16 h[8]; };

static __device__ inline v8f wmma16(const v16h& a, const v16h& b, const v8f& c) {
    return __builtin_amdgcn_wmma_f32_16x16x32_f16(false, a, false, b, (short)0, c, false, false);
}

// Async 16-byte global -> LDS copy (per lane). VDST = LDS byte address
// (low 32 bits of the flat shared-aperture pointer), VADDR = 64-bit global.
static __device__ inline void async_ld16(void* lds_ptr, const void* gptr) {
    unsigned lds_off = (unsigned)(uintptr_t)lds_ptr;
    asm volatile("global_load_async_to_lds_b128 %0, %1, off"
                 :: "v"(lds_off), "v"(gptr) : "memory");
}
static __device__ inline void wait_async0() {
    asm volatile("s_wait_asynccnt 0x0" ::: "memory");
}

// ---------------------------------------------------------------------------
// Tiled GEMM: C[M,N] = A[M,K] * W[N,K]^T
// AF16: 0 -> A is f32 (convert while staging), 1 -> A is f16 (async staging)
// MODE: 0 -> f16 out row-major; 1 -> f16 out transposed [b][N][2048] (V proj);
//       2 -> f32 out + bias (final projection)
// Block: 256 threads (8 waves), 128x128 output tile, K-tile = 64.
// ---------------------------------------------------------------------------
template <int AF16, int MODE>
__global__ __launch_bounds__(256)
void gemm_kernel(const void* __restrict__ Aptr, const float* __restrict__ W,
                 const float* __restrict__ bias, void* __restrict__ Out,
                 int M, int N, int K) {
    __shared__ __align__(16) _Float16 As[128][72];
    __shared__ __align__(16) _Float16 Ws[128][72];

    const int t     = threadIdx.x;
    const int lane  = t & 31;
    const int w     = t >> 5;
    const int ln    = lane & 15;
    const int hi    = lane >> 4;
    const int waveM = w & 3;     // 4 waves along M, 32 rows each
    const int waveN = w >> 2;    // 2 waves along N, 64 cols each
    const int mbase = blockIdx.y * 128;
    const int nbase = blockIdx.x * 128;

    v8f acc[2][4];
#pragma unroll
    for (int i = 0; i < 2; ++i)
#pragma unroll
        for (int j = 0; j < 4; ++j)
            acc[i][j] = (v8f){0.f,0.f,0.f,0.f,0.f,0.f,0.f,0.f};

    const int r   = t >> 1;
    const int seg = (t & 1) * 32;

    for (int kb = 0; kb < K; kb += 64) {
        __syncthreads();
        // ---- stage A tile (128 x 64) as f16 ----
        if (AF16) {
            const _Float16* src = (const _Float16*)Aptr + (size_t)(mbase + r) * K + kb + seg;
#pragma unroll
            for (int i = 0; i < 4; ++i)
                async_ld16(&As[r][seg + 8*i], src + 8*i);
        } else {
            const float4* src = (const float4*)((const float*)Aptr +
                                (size_t)(mbase + r) * K + kb + seg);
#pragma unroll
            for (int i = 0; i < 4; ++i) {
                float4 a0 = src[2*i], a1 = src[2*i+1];
                H8 x;
                x.h[0]=(_Float16)a0.x; x.h[1]=(_Float16)a0.y;
                x.h[2]=(_Float16)a0.z; x.h[3]=(_Float16)a0.w;
                x.h[4]=(_Float16)a1.x; x.h[5]=(_Float16)a1.y;
                x.h[6]=(_Float16)a1.z; x.h[7]=(_Float16)a1.w;
                *(uint4*)&As[r][seg + 8*i] = x.u;
            }
        }
        // ---- stage W tile (128 x 64) as f16 (convert f32 -> f16) ----
        {
            const float4* src = (const float4*)(W + (size_t)(nbase + r) * K + kb + seg);
#pragma unroll
            for (int i = 0; i < 4; ++i) {
                float4 a0 = src[2*i], a1 = src[2*i+1];
                H8 x;
                x.h[0]=(_Float16)a0.x; x.h[1]=(_Float16)a0.y;
                x.h[2]=(_Float16)a0.z; x.h[3]=(_Float16)a0.w;
                x.h[4]=(_Float16)a1.x; x.h[5]=(_Float16)a1.y;
                x.h[6]=(_Float16)a1.z; x.h[7]=(_Float16)a1.w;
                *(uint4*)&Ws[r][seg + 8*i] = x.u;
            }
        }
        if (AF16) wait_async0();
        __syncthreads();

#pragma unroll
        for (int kk = 0; kk < 64; kk += 32) {
            Frag a[2], b[4];
#pragma unroll
            for (int tm = 0; tm < 2; ++tm) {
                const _Float16* base = &As[waveM*32 + tm*16 + ln][kk + hi*8];
                a[tm].u[0] = *(const uint4*)base;        // K = kk + hi*8 .. +7
                a[tm].u[1] = *(const uint4*)(base + 16); // K = kk+16+hi*8 .. +7
            }
#pragma unroll
            for (int tn = 0; tn < 4; ++tn) {
                const _Float16* base = &Ws[waveN*64 + tn*16 + ln][kk + hi*8];
                b[tn].u[0] = *(const uint4*)base;        // K = kk + hi*8 .. +7
                b[tn].u[1] = *(const uint4*)(base + 16);
            }
#pragma unroll
            for (int tm = 0; tm < 2; ++tm)
#pragma unroll
                for (int tn = 0; tn < 4; ++tn)
                    acc[tm][tn] = wmma16(a[tm].h, b[tn].h, acc[tm][tn]);
        }
    }

    // ---- epilogue ----
#pragma unroll
    for (int tm = 0; tm < 2; ++tm)
#pragma unroll
        for (int tn = 0; tn < 4; ++tn)
#pragma unroll
            for (int j = 0; j < 8; ++j) {
                int row = mbase + waveM*32 + tm*16 + j + 8*hi;
                int col = nbase + waveN*64 + tn*16 + ln;
                float v = acc[tm][tn][j];
                if (MODE == 0) {
                    ((_Float16*)Out)[(size_t)row * N + col] = (_Float16)v;
                } else if (MODE == 1) {
                    int bidx = row >> 11;       // n = 2048 tokens per batch
                    int tok  = row & 2047;
                    ((_Float16*)Out)[(size_t)bidx * N * 2048 + (size_t)col * 2048 + tok] =
                        (_Float16)v;
                } else {
                    ((float*)Out)[(size_t)row * N + col] = v + bias[col];
                }
            }
}

// ---------------------------------------------------------------------------
// Flash attention: one block per (q-block of 128 rows, head, batch).
// 8 waves; each wave owns 16 query rows. d_k = 64, key blocks of 128.
// K/V tiles double-buffered, staged with async global->LDS copies.
// ---------------------------------------------------------------------------
__global__ __launch_bounds__(256)
void attn_kernel(const _Float16* __restrict__ qp,  // [4096, 2048]
                 const _Float16* __restrict__ kp,  // [4096, 512]
                 const _Float16* __restrict__ vT,  // [2, 512, 2048]
                 _Float16* __restrict__ op) {      // [4096, 2048]
    __shared__ __align__(16) _Float16 Ks[2][128][72];   // K block [key][d]
    __shared__ __align__(16) _Float16 Vt[2][64][136];   // V block [d][key]
    __shared__ __align__(16) _Float16 Ps[8][16][136];   // per-wave P tile

    const int t    = threadIdx.x;
    const int lane = t & 31;
    const int w    = t >> 5;
    const int ln   = lane & 15;
    const int hi   = lane >> 4;
    const int qb   = blockIdx.x;   // 0..15
    const int h    = blockIdx.y;   // 0..31
    const int bb   = blockIdx.z;   // 0..1
    const int kv   = h >> 2;
    const float scale = 0.125f;    // 1/sqrt(64)

    // ---- stage this block's Q rows (128 x 64) into Ks[0], grab resident frags ----
    {
        int r = t >> 1, seg = (t & 1) * 32;
        const _Float16* src = qp + (size_t)(bb*2048 + qb*128 + r) * 2048 + h*64 + seg;
#pragma unroll
        for (int i = 0; i < 4; ++i)
            async_ld16(&Ks[0][r][seg + 8*i], src + 8*i);
    }
    wait_async0();
    __syncthreads();
    Frag qf[2];
#pragma unroll
    for (int kk = 0; kk < 2; ++kk) {
        const _Float16* base = &Ks[0][w*16 + ln][kk*32 + hi*8];
        qf[kk].u[0] = *(const uint4*)base;
        qf[kk].u[1] = *(const uint4*)(base + 16);
    }
    __syncthreads();   // everyone has their Q frags before Ks[0] is reused

    v8f oacc[4];
#pragma unroll
    for (int i = 0; i < 4; ++i) oacc[i] = (v8f){0.f,0.f,0.f,0.f,0.f,0.f,0.f,0.f};
    float m[8], l[8];
#pragma unroll
    for (int j = 0; j < 8; ++j) { m[j] = -1e30f; l[j] = 0.f; }

    // async stage of K/V block jb into buffer buf
    auto stage_kv = [&](int jb, int buf) {
        int r = t >> 1, seg = (t & 1) * 32;
        const _Float16* ksrc = kp + (size_t)(bb*2048 + jb*128 + r) * 512 + kv*64 + seg;
#pragma unroll
        for (int i = 0; i < 4; ++i)
            async_ld16(&Ks[buf][r][seg + 8*i], ksrc + 8*i);
        int d = t >> 2, sg = (t & 3) * 32;
        const _Float16* vsrc = vT + (size_t)bb*512*2048 +
                               (size_t)(kv*64 + d) * 2048 + jb*128 + sg;
#pragma unroll
        for (int i = 0; i < 4; ++i)
            async_ld16(&Vt[buf][d][sg + 8*i], vsrc + 8*i);
    };

    stage_kv(0, 0);   // prefetch first block

    for (int jb = 0; jb < 16; ++jb) {
        const int cur = jb & 1;
        wait_async0();      // this wave's copies for block jb are in LDS
        __syncthreads();    // ...and everyone else's too
        if (jb < 15) stage_kv(jb + 1, cur ^ 1);   // prefetch next (no wait)

        // ---- S = Q * K^T (16 x 128) ----
        v8f sacc[8];
#pragma unroll
        for (int s = 0; s < 8; ++s) sacc[s] = (v8f){0.f,0.f,0.f,0.f,0.f,0.f,0.f,0.f};
#pragma unroll
        for (int kk = 0; kk < 2; ++kk)
#pragma unroll
            for (int s = 0; s < 8; ++s) {
                Frag bf;
                const _Float16* base = &Ks[cur][s*16 + ln][kk*32 + hi*8];
                bf.u[0] = *(const uint4*)base;
                bf.u[1] = *(const uint4*)(base + 16);
                sacc[s] = wmma16(qf[kk].h, bf.h, sacc[s]);
            }

        // ---- online softmax (rows live in 16-lane halves) ----
#pragma unroll
        for (int j = 0; j < 8; ++j) {
            float rmax = -1e30f;
#pragma unroll
            for (int s = 0; s < 8; ++s) rmax = fmaxf(rmax, sacc[s][j]);
#pragma unroll
            for (int off = 1; off < 16; off <<= 1)
                rmax = fmaxf(rmax, __shfl_xor(rmax, off, 32));
            float mnew  = fmaxf(m[j], rmax * scale);
            float alpha = __expf(m[j] - mnew);
            float rs = 0.f;
#pragma unroll
            for (int s = 0; s < 8; ++s) {
                float p = __expf(sacc[s][j] * scale - mnew);
                rs += p;
                Ps[w][j + 8*hi][s*16 + ln] = (_Float16)p;
            }
#pragma unroll
            for (int off = 1; off < 16; off <<= 1)
                rs += __shfl_xor(rs, off, 32);
            l[j] = l[j] * alpha + rs;
            m[j] = mnew;
#pragma unroll
            for (int tt = 0; tt < 4; ++tt) oacc[tt][j] *= alpha;
        }

        // ---- O += P * V  (same-wave LDS store->load is in order; no barrier) ----
#pragma unroll
        for (int kk = 0; kk < 4; ++kk) {
            Frag pa;
            const _Float16* pb = &Ps[w][ln][kk*32 + hi*8];
            pa.u[0] = *(const uint4*)pb;
            pa.u[1] = *(const uint4*)(pb + 16);
#pragma unroll
            for (int tt = 0; tt < 4; ++tt) {
                Frag vf;
                const _Float16* vb = &Vt[cur][tt*16 + ln][kk*32 + hi*8];
                vf.u[0] = *(const uint4*)vb;
                vf.u[1] = *(const uint4*)(vb + 16);
                oacc[tt] = wmma16(pa.h, vf.h, oacc[tt]);
            }
        }
    }

    // ---- epilogue: O /= l, write f16 ----
#pragma unroll
    for (int j = 0; j < 8; ++j) {
        float inv = 1.0f / l[j];
        int row = bb*2048 + qb*128 + w*16 + j + 8*hi;
#pragma unroll
        for (int tt = 0; tt < 4; ++tt) {
            int col = h*64 + tt*16 + ln;
            op[(size_t)row * 2048 + col] = (_Float16)(oacc[tt][j] * inv);
        }
    }
}

// ---------------------------------------------------------------------------
extern "C" void kernel_launch(void* const* d_in, const int* in_sizes, int n_in,
                              void* d_out, int out_size, void* d_ws, size_t ws_size,
                              hipStream_t stream) {
    const float* Q   = (const float*)d_in[0];
    const float* K   = (const float*)d_in[1];
    const float* V   = (const float*)d_in[2];
    const float* w_q = (const float*)d_in[3];
    const float* w_k = (const float*)d_in[4];
    const float* w_v = (const float*)d_in[5];
    const float* w_o = (const float*)d_in[6];
    const float* b_o = (const float*)d_in[7];
    float* out = (float*)d_out;

    char* ws = (char*)d_ws;
    const size_t SZ_Q = 4096ull * 2048 * 2;  // 16 MB  f16
    const size_t SZ_K = 4096ull * 512 * 2;   //  4 MB  f16
    _Float16* qproj = (_Float16*)(ws);
    _Float16* kproj = (_Float16*)(ws + SZ_Q);
    _Float16* vTr   = (_Float16*)(ws + SZ_Q + SZ_K);
    _Float16* oat   = (_Float16*)(ws + SZ_Q + 2 * SZ_K);

    dim3 blk(256);
    // projections (A = f32 activations, W = f32 weights)
    gemm_kernel<0, 0><<<dim3(16, 32), blk, 0, stream>>>(Q, w_q, nullptr, qproj, 4096, 2048, 2048);
    gemm_kernel<0, 0><<<dim3( 4, 32), blk, 0, stream>>>(K, w_k, nullptr, kproj, 4096,  512, 2048);
    gemm_kernel<0, 1><<<dim3( 4, 32), blk, 0, stream>>>(V, w_v, nullptr, vTr,   4096,  512, 2048);
    // attention
    attn_kernel<<<dim3(16, 32, 2), blk, 0, stream>>>(qproj, kproj, vTr, oat);
    // output projection + bias (A = f16 attention output)
    gemm_kernel<1, 2><<<dim3(16, 32), blk, 0, stream>>>(oat, w_o, b_o, out, 4096, 2048, 2048);
}
